// YunaAttention_14156212207974
// MI455X (gfx1250) — matrix operand order repaired and checked by reading, not compiled, not run on hardware
//
#include <hip/hip_runtime.h>
#include <hip/hip_bf16.h>

// ---------------------------------------------------------------------------
// CDNA5 (gfx1250) fused attention block: bf16 WMMA GEMMs + flash attention.
// Wave32 only. WMMA shape 16x16x32 (bf16 in, f32 acc).
// ---------------------------------------------------------------------------

typedef __attribute__((ext_vector_type(16))) __bf16 v16bf;
typedef __attribute__((ext_vector_type(8)))  __bf16 v8bf;
typedef __attribute__((ext_vector_type(8)))  float  v8f;

// --- WMMA fragment loaders (per ISA 7.12.2 register layouts) ---------------
// A (16x32, 16-bit): lane l (row M=l%16), half=l/16:
//   elements 0..7  = K [k0 + 8*half  .. +7]
//   elements 8..15 = K [k0 + 16 + 8*half .. +7]
// B (32x16, 16-bit): lane l -> column N=l%16:
//   elements 0..7  = K [k0 + 8*half .. +7], elements 8..15 = K [k0 + 16 + 8*half .. +7]
static __device__ __forceinline__ v16bf frag_combine(v8bf lo, v8bf hi) {
  return __builtin_shufflevector(lo, hi, 0, 1, 2, 3, 4, 5, 6, 7,
                                 8, 9, 10, 11, 12, 13, 14, 15);
}
static __device__ __forceinline__ v16bf load_frag_A(const __bf16* __restrict__ row, int k0, int hh) {
  v8bf lo = *(const v8bf*)(row + k0 + hh * 8);
  v8bf hi = *(const v8bf*)(row + k0 + 16 + hh * 8);
  return frag_combine(lo, hi);
}
static __device__ __forceinline__ v16bf load_frag_B(const __bf16* __restrict__ row, int k0, int hh) {
  v8bf lo = *(const v8bf*)(row + k0 + hh * 8);
  v8bf hi = *(const v8bf*)(row + k0 + 16 + hh * 8);
  return frag_combine(lo, hi);
}

static __device__ __forceinline__ v8f wmma_bf16(v16bf a, v16bf b, v8f c) {
  return __builtin_amdgcn_wmma_f32_16x16x32_bf16(false, a, false, b, (short)0, c, false, false);
}

// raw max (skips clang's quiet-NaN canonicalize; inputs are never sNaN here)
static __device__ __forceinline__ float fmax_raw(float a, float b) {
  float d;
  asm("v_max_num_f32 %0, %1, %2" : "=v"(d) : "v"(a), "v"(b));
  return d;
}

// --- DPP16 butterfly reductions within each 16-lane row (pure VALU) ---------
// stages: xor1 = quad_perm(1,0,3,2)=0xB1, xor2 = quad_perm(2,3,0,1)=0x4E,
//         cross-quad = row_half_mirror(0x141), cross-8 = row_mirror(0x140).
template <int CTRL>
static __device__ __forceinline__ float dpp_movf(float v) {
  return __int_as_float(
      __builtin_amdgcn_update_dpp(0, __float_as_int(v), CTRL, 0xF, 0xF, true));
}
static __device__ __forceinline__ float red_max16(float v) {
  v = fmax_raw(v, dpp_movf<0xB1>(v));
  v = fmax_raw(v, dpp_movf<0x4E>(v));
  v = fmax_raw(v, dpp_movf<0x141>(v));
  v = fmax_raw(v, dpp_movf<0x140>(v));
  return v;
}
static __device__ __forceinline__ float red_sum16(float v) {
  v += dpp_movf<0xB1>(v);
  v += dpp_movf<0x4E>(v);
  v += dpp_movf<0x141>(v);
  v += dpp_movf<0x140>(v);
  return v;
}

// --- fp32 -> bf16 conversion ------------------------------------------------
__global__ void cvt_f32_bf16(const float* __restrict__ in, __bf16* __restrict__ out, size_t n) {
  size_t i = (size_t)blockIdx.x * blockDim.x + threadIdx.x;
  size_t stride = (size_t)gridDim.x * blockDim.x;
  for (; i < n; i += stride) out[i] = (__bf16)in[i];
}

// --- GEMM: C[M,N] = A[M,K](bf16) * W[N,K]^T(bf16) + bias --------------------
// Register-tiled: each wave computes MT x NT 16x16 tiles (64x32 for 4x2).
// 4 waves per block -> block covers (4*MT*16) x (NT*16). K%32==0.
template <typename OutT, int MT, int NT>
__global__ __launch_bounds__(128)
void gemm_wmma(const __bf16* __restrict__ A, const __bf16* __restrict__ W,
               const float* __restrict__ bias, OutT* __restrict__ C,
               int M, int N, int K) {
  const int lane = threadIdx.x & 31;
  const int wv   = threadIdx.x >> 5;
  const int hh   = lane >> 4;
  const int l16  = lane & 15;
  const int nBase = blockIdx.x * (NT * 16);
  const int mBase = (blockIdx.y * 4 + wv) * (MT * 16);

  const __bf16* aPtr = A + (size_t)(mBase + l16) * K;   // +mt*16*K (imm-foldable)
  const __bf16* bPtr = W + (size_t)(nBase + l16) * K;   // +nt*16*K

  v8f acc[MT][NT] = {};
  for (int k0 = 0; k0 < K; k0 += 32) {
    v16bf af[MT], bf[NT];
#pragma unroll
    for (int nt = 0; nt < NT; ++nt)
      bf[nt] = load_frag_B(bPtr + (size_t)nt * 16 * K, k0, hh);
#pragma unroll
    for (int mt = 0; mt < MT; ++mt)
      af[mt] = load_frag_A(aPtr + (size_t)mt * 16 * K, k0, hh);
#pragma unroll
    for (int mt = 0; mt < MT; ++mt)
#pragma unroll
      for (int nt = 0; nt < NT; ++nt)
        acc[mt][nt] = wmma_bf16(af[mt], bf[nt], acc[mt][nt]);
  }
#pragma unroll
  for (int nt = 0; nt < NT; ++nt) {
    const int col = nBase + nt * 16 + l16;
    const float bv = bias ? bias[col] : 0.0f;
#pragma unroll
    for (int mt = 0; mt < MT; ++mt) {
      const int row0 = mBase + mt * 16 + hh * 8;
#pragma unroll
      for (int r = 0; r < 8; ++r)
        C[(size_t)(row0 + r) * N + col] = (OutT)(acc[mt][nt][r] + bv);
    }
  }
}

// --- multi-section rotary + repack to [B,H,T,D] bf16 ------------------------
// sections [16,24,24,16,24,24] pick modality (i%3) of cos/sin[3,B,T,D].
__global__ void rope_pack(const __bf16* __restrict__ qin, const float* __restrict__ cosb,
                          const float* __restrict__ sinb, __bf16* __restrict__ qout,
                          int Bn, int T, int H, int D) {
  const int d = threadIdx.x;                 // 0..127
  const int t = blockIdx.x % T;
  const int h = (blockIdx.x / T) % H;
  const int b = blockIdx.x / (T * H);
  const int m = (d < 16) ? 0 : (d < 40) ? 1 : (d < 64) ? 2
              : (d < 80) ? 0 : (d < 104) ? 1 : 2;
  const size_t ci = (((size_t)m * Bn + b) * T + t) * D + d;
  const float c = cosb[ci];
  const float s = sinb[ci];
  const size_t inBase = (((size_t)b * T + t) * H + h) * (size_t)D;
  const float q = (float)qin[inBase + d];
  const float o = (d < 64) ? -(float)qin[inBase + d + 64]
                           :  (float)qin[inBase + d - 64];
  qout[(((size_t)b * H + h) * T + t) * D + d] = (__bf16)(q * c + o * s);
}

// --- V repack (transpose) to [B,KV,D,T] bf16 --------------------------------
__global__ void v_pack(const __bf16* __restrict__ vin, __bf16* __restrict__ vout,
                       int Bn, int KV, int T, int D) {
  const int d  = blockIdx.x % D;
  const int kv = (blockIdx.x / D) % KV;
  const int b  = blockIdx.x / (D * KV);
  for (int t = threadIdx.x; t < T; t += blockDim.x)
    vout[(((size_t)b * KV + kv) * D + d) * T + t] =
        vin[(((size_t)b * T + t) * KV + kv) * D + d];
}

// --- flash attention: one wave per (b, h, 16-query tile) --------------------
// Q [B,H,T,D], K [B,KV,T,D], Vt [B,KV,D,T], out Y [B,T,H,D] bf16.
__global__ __launch_bounds__(128)
void flash_attn(const __bf16* __restrict__ Q, const __bf16* __restrict__ K,
                const __bf16* __restrict__ Vt, __bf16* __restrict__ Y,
                int Bn, int H, int KV, int T, int D) {
  __shared__ __bf16 pS[4][16 * 32];          // 1KB per wave: P in row-major
  const int lane = threadIdx.x & 31;
  const int wv   = threadIdx.x >> 5;
  const int hh   = lane >> 4;
  const int l16  = lane & 15;
  const int qtPer = T / 16;
  const int gw = blockIdx.x * 4 + wv;
  const int qt = gw % qtPer;
  const int h  = (gw / qtPer) % H;
  const int b  = gw / qtPer / H;
  const int kvh = h / (H / KV);
  const int qBase = qt * 16;
  const float SCALE = 0.08838834764831845f;  // 1/sqrt(128)

  // Q A-fragments covering D=128 (4 fragments of 16x32)
  const __bf16* qRow = Q + ((size_t)((b * H + h) * T) + qBase + l16) * D;
  v16bf qf[4];
#pragma unroll
  for (int f = 0; f < 4; ++f) qf[f] = load_frag_A(qRow, f * 32, hh);

  v8f o[8] = {};                             // O accumulators, 8 d-tiles of 16
  float rmax[8], rsum[8];
#pragma unroll
  for (int r = 0; r < 8; ++r) { rmax[r] = -1e30f; rsum[r] = 0.0f; }

  // per-lane streaming pointers (advance by 32 keys per block)
  const __bf16* kCur = K  + ((size_t)((b * KV + kvh) * T) + l16) * D;      // key l16
  const __bf16* vCur = Vt + ((size_t)((b * KV + kvh) * D) + l16) * T;      // dim l16, +j*16*T imm

  const int nKB = (qBase + 16 + 31) / 32;    // causal: key blocks of 32
  for (int kb32 = 0; kb32 < nKB; ++kb32) {
    const int kb = kb32 * 32;
    // prefetch next key block rows toward WGP (global_prefetch_b8)
    __builtin_prefetch(kCur + (size_t)32 * D, 0, 3);
    __builtin_prefetch(kCur + (size_t)48 * D, 0, 3);

    // ---- load ALL 8 K fragments first (distinct live ranges -> 16 b128 in flight)
    v16bf kf[8];
#pragma unroll
    for (int f = 0; f < 4; ++f) {
      kf[f]     = load_frag_B(kCur, f * 32, hh);
      kf[4 + f] = load_frag_B(kCur + (size_t)16 * D, f * 32, hh);
    }
    // S = Q * K^T  (two 16x16 tiles over 32 keys, summed over D)
    v8f s1 = {}, s2 = {};
#pragma unroll
    for (int f = 0; f < 4; ++f) {
      s1 = wmma_bf16(qf[f], kf[f], s1);
      s2 = wmma_bf16(qf[f], kf[4 + f], s2);
    }

    // ---- issue V fragment loads NOW so they overlap the softmax VALU chain
    v16bf vf[8];
#pragma unroll
    for (int j = 0; j < 8; ++j)
      vf[j] = load_frag_B(vCur + (size_t)j * 16 * T, 0, hh);

    // online softmax per row (row = r + 8*hh in C-fragment layout)
#pragma unroll
    for (int r = 0; r < 8; ++r) {
      const int qi = qBase + r + 8 * hh;
      float a = s1[r] * SCALE; if (kb + l16 > qi)      a = -1e30f;
      float c = s2[r] * SCALE; if (kb + 16 + l16 > qi) c = -1e30f;
      const float bm = red_max16(fmax_raw(a, c));
      const float nm = fmax_raw(rmax[r], bm);
      const float alpha = __expf(rmax[r] - nm);
      rmax[r] = nm;
      const float p1 = __expf(a - nm);
      const float p2 = __expf(c - nm);
      rsum[r] = rsum[r] * alpha + red_sum16(p1 + p2);
#pragma unroll
      for (int j = 0; j < 8; ++j) o[j][r] *= alpha;
      const int row = r + 8 * hh;
      pS[wv][row * 32 + l16]      = (__bf16)p1;
      pS[wv][row * 32 + 16 + l16] = (__bf16)p2;
    }
    asm volatile("s_wait_dscnt 0" ::: "memory");   // LDS stores visible to our loads
    // P A-fragment (16x32) from LDS (transposes C-layout -> A-layout)
    v16bf pf;
    {
      const __bf16* pr = &pS[wv][l16 * 32];
      v8bf lo = *(const v8bf*)(pr + hh * 8);
      v8bf hi = *(const v8bf*)(pr + 16 + hh * 8);
      pf = frag_combine(lo, hi);
    }
    // O += P * V  (8 d-tiles of 16)
#pragma unroll
    for (int j = 0; j < 8; ++j)
      o[j] = wmma_bf16(pf, vf[j], o[j]);

    kCur += (size_t)32 * D;
    vCur += 32;
  }

  // normalize + write Y[B,T,H,D] bf16 (input to the O-projection GEMM)
#pragma unroll
  for (int r = 0; r < 8; ++r) {
    const int qg = qBase + r + 8 * hh;
    const float inv = 1.0f / rsum[r];
    const size_t base = (((size_t)b * T + qg) * H + h) * (size_t)D;
#pragma unroll
    for (int j = 0; j < 8; ++j)
      Y[base + j * 16 + l16] = (__bf16)(o[j][r] * inv);
  }
}

// ---------------------------------------------------------------------------
extern "C" void kernel_launch(void* const* d_in, const int* in_sizes, int n_in,
                              void* d_out, int out_size, void* d_ws, size_t ws_size,
                              hipStream_t stream) {
  (void)in_sizes; (void)n_in; (void)out_size; (void)ws_size;
  const float* x    = (const float*)d_in[0];
  const float* cosb = (const float*)d_in[1];
  const float* sinb = (const float*)d_in[2];
  const float* q_w  = (const float*)d_in[3];
  const float* q_b  = (const float*)d_in[4];
  const float* k_w  = (const float*)d_in[5];
  const float* k_b  = (const float*)d_in[6];
  const float* v_w  = (const float*)d_in[7];
  const float* v_b  = (const float*)d_in[8];
  const float* o_w  = (const float*)d_in[9];
  float* out = (float*)d_out;

  const int Bn = 2, T = 2048, E = 4096, H = 32, D = 128, KV = 8, KVE = 1024;
  const int M = Bn * T;

  char* ws = (char*)d_ws;
  size_t off = 0;
  auto take = [&](size_t bytes) -> char* {
    char* p = ws + off; off += (bytes + 255) & ~(size_t)255; return p;
  };
  __bf16* xb  = (__bf16*)take((size_t)M * E * 2);     // x bf16       [M,E]
  __bf16* wq  = (__bf16*)take((size_t)E * E * 2);     // q_w bf16     [E,E]
  __bf16* wk  = (__bf16*)take((size_t)KVE * E * 2);   // k_w bf16     [KVE,E]
  __bf16* wvv = (__bf16*)take((size_t)KVE * E * 2);   // v_w bf16
  __bf16* wo  = (__bf16*)take((size_t)E * E * 2);     // o_w bf16
  __bf16* qr  = (__bf16*)take((size_t)M * E * 2);     // q raw        [M,E]
  __bf16* kr  = (__bf16*)take((size_t)M * KVE * 2);   // k raw        [M,KVE]
  __bf16* vr  = (__bf16*)take((size_t)M * KVE * 2);   // v raw
  __bf16* qp  = (__bf16*)take((size_t)M * E * 2);     // q roped      [B,H,T,D]
  __bf16* kp  = (__bf16*)take((size_t)M * KVE * 2);   // k roped      [B,KV,T,D]
  __bf16* vt  = (__bf16*)take((size_t)M * KVE * 2);   // v transposed [B,KV,D,T]
  __bf16* yb  = (__bf16*)take((size_t)M * E * 2);     // attn out     [M,E]

  // 1) fp32 -> bf16
  cvt_f32_bf16<<<2048, 256, 0, stream>>>(x,   xb,  (size_t)M * E);
  cvt_f32_bf16<<<2048, 256, 0, stream>>>(q_w, wq,  (size_t)E * E);
  cvt_f32_bf16<<<1024, 256, 0, stream>>>(k_w, wk,  (size_t)KVE * E);
  cvt_f32_bf16<<<1024, 256, 0, stream>>>(v_w, wvv, (size_t)KVE * E);
  cvt_f32_bf16<<<2048, 256, 0, stream>>>(o_w, wo,  (size_t)E * E);

  // 2) projections: y = x @ W^T + b  (WMMA, 64x32 per wave)
  gemm_wmma<__bf16, 4, 2><<<dim3(E / 32,   M / 256), 128, 0, stream>>>(xb, wq,  q_b, qr, M, E,   E);
  gemm_wmma<__bf16, 4, 2><<<dim3(KVE / 32, M / 256), 128, 0, stream>>>(xb, wk,  k_b, kr, M, KVE, E);
  gemm_wmma<__bf16, 4, 2><<<dim3(KVE / 32, M / 256), 128, 0, stream>>>(xb, wvv, v_b, vr, M, KVE, E);

  // 3) rotary + layout repack
  rope_pack<<<Bn * H * T,  D, 0, stream>>>(qr, cosb, sinb, qp, Bn, T, H,  D);
  rope_pack<<<Bn * KV * T, D, 0, stream>>>(kr, cosb, sinb, kp, Bn, T, KV, D);
  v_pack<<<Bn * KV * D, 256, 0, stream>>>(vr, vt, Bn, KV, T, D);

  // 4) flash attention (causal, GQA): 1 wave per 16-query tile
  flash_attn<<<(Bn * H * (T / 16)) / 4, 128, 0, stream>>>(qp, kp, vt, yb, Bn, H, KV, T, D);

  // 5) output projection -> fp32 d_out
  gemm_wmma<float, 4, 2><<<dim3(E / 32, M / 256), 128, 0, stream>>>(yb, wo, nullptr, out, M, E, E);
}
